// SAGEConv_22351009809096
// MI455X (gfx1250) — matrix operand order repaired and checked by reading, not compiled, not run on hardware
//
#include <hip/hip_runtime.h>
#include <hip/hip_bf16.h>
#include <cstdint>

#define L_ 8
#define G_ 50000
#define CIN_ 128
#define COUT_ 128
#define M_ 32
#define E_ 800000
#define CAT_ 192
#define N_ (L_*G_)     // 400000 rows
#define EPS_ 1e-5f

typedef __bf16 v16bf __attribute__((ext_vector_type(16)));
typedef float  v8f   __attribute__((ext_vector_type(8)));
typedef unsigned int u32x4 __attribute__((ext_vector_type(4)));
typedef int          i32x8 __attribute__((ext_vector_type(8)));
typedef int          i32x4 __attribute__((ext_vector_type(4)));

struct U32x8 { uint4 lo, hi; };

__device__ __forceinline__ v16bf make_frag(uint4 lo, uint4 hi) {
  U32x8 t{lo, hi};
  return __builtin_bit_cast(v16bf, t);
}

__device__ __forceinline__ float silu_f(float x) { return x / (1.0f + __expf(-x)); }

// LDS byte offset from a __shared__ pointer: flat LDS aperture keeps the
// wave-relative LDS address in the low 32 bits.
__device__ __forceinline__ uint32_t lds_off(const void* p) {
  return (uint32_t)(uintptr_t)p;
}

// CDNA5 async global->LDS copy, 16 bytes per lane (tracked by ASYNCcnt).
__device__ __forceinline__ void async_b128(uint32_t ldsoff, const void* g) {
  asm volatile("global_load_async_to_lds_b128 %0, %1, off"
               :: "v"(ldsoff), "v"(g) : "memory");
}
__device__ __forceinline__ void wait_async0() {
  asm volatile("s_wait_asynccnt 0x0" ::: "memory");
}

// CDNA5 Tensor Data Mover: one descriptor DMAs `nbytes` (multiple of 8,
// <= 512KB) from global to LDS. D# per ISA ch.8: 1-row tensor of 8-byte
// elements, tile_dim0 = element count. Issued per-wave (EXEC ignored),
// tracked by TENSORcnt. This toolchain's builtin takes 6 args:
// (u32x4 g0, i32x8 g1, i32x4 g2, i32x4 g3, i32x8 pad, i32 cpol).
__device__ __forceinline__ void tdm_load_to_lds(uint32_t lds_addr, const void* gptr,
                                                uint32_t nbytes) {
#if __has_builtin(__builtin_amdgcn_tensor_load_to_lds)
  const uint64_t ga = (uint64_t)(uintptr_t)gptr;
  const uint32_t elems = nbytes >> 3;               // 8-byte elements
  u32x4 g0;
  g0[0] = 1u;                                       // count=1, user descriptor
  g0[1] = lds_addr;                                 // lds_addr [63:32]
  g0[2] = (uint32_t)ga;                             // global_addr low
  g0[3] = (uint32_t)(ga >> 32) | (2u << 30);        // global_addr high | type=2
  i32x8 g1;
  g1[0] = (int)(3u << 16);                          // data_size=8B, mask=0
  g1[1] = (int)((elems & 0xFFFFu) << 16);           // tensor_dim0[15:0]
  g1[2] = (int)((elems >> 16) | (1u << 16));        // tensor_dim0[31:16], tensor_dim1=1
  g1[3] = (int)((elems & 0xFFFFu) << 16);           // tile_dim0
  g1[4] = 1;                                        // tile_dim1=1, tile_dim2=0
  g1[5] = (int)elems;                               // tensor_dim0_stride[31:0]
  g1[6] = 0;
  g1[7] = 0;
  i32x4 z4 = {};
  i32x8 z8 = {};
  __builtin_amdgcn_tensor_load_to_lds(g0, g1, z4, z4, z8, 0);
  __builtin_amdgcn_s_wait_tensorcnt(0);
#else
  // fallback: async copies spread over the wave
  const int lane = (int)(threadIdx.x & 31);
  for (uint32_t o = (uint32_t)lane * 16; o < nbytes; o += 32 * 16)
    async_b128(lds_addr + o, (const char*)gptr + o);
  wait_async0();
#endif
}

// ---------------- K0: weight conversion fp32 -> bf16 ----------------
__global__ void k0_convert(const float* __restrict__ Wlin, const float* __restrict__ Wain,
                           const float* __restrict__ Waout, const float* __restrict__ Wmlp,
                           __bf16* __restrict__ Wcat, __bf16* __restrict__ WmlpB) {
  int i = blockIdx.x * blockDim.x + threadIdx.x;
  if (i < CIN_ * CAT_) {
    int k = i / CAT_, c = i % CAT_;
    float v;
    if (c < COUT_)            v = Wlin[k * COUT_ + c];
    else if (c < COUT_ + M_)  v = Wain[k * M_ + (c - COUT_)];
    else                      v = Waout[k * M_ + (c - COUT_ - M_)];
    Wcat[c * CIN_ + k] = (__bf16)v;
  }
  if (i < CAT_ * COUT_) WmlpB[(i % COUT_) * CAT_ + i / COUT_] = (__bf16)Wmlp[i];
}

// ---------------- K1: fused input GEMM  x[N,128] @ Wcat[128,192] ----------------
// block: 256 threads = 8 waves, tile 128 rows x 192 cols, K = 128 in 4 steps of 32
__global__ __launch_bounds__(256, 1) void k1_gemm_in(
    const float* __restrict__ x, const __bf16* __restrict__ Wcat,
    const float* __restrict__ b_lin, const float* __restrict__ b_ain,
    const float* __restrict__ b_aout,
    float* __restrict__ hcat, float* __restrict__ Pin, float* __restrict__ Pout) {
  __shared__ alignas(16) __bf16 ldsA[128 * CIN_];   // 32 KB
  __shared__ alignas(16) __bf16 ldsB[CIN_ * CAT_];  // 48 KB
  const int tid = threadIdx.x;
  const int rowBase = blockIdx.x * 128;

  // weights: async copy straight into LDS (bf16, byte-for-byte)
  {
    const uint32_t bbase = lds_off(&ldsB[0]);
    for (int it = tid; it < CIN_ * CAT_ / 8; it += 256)
      async_b128(bbase + it * 16, (const char*)Wcat + it * 16);
  }
  // x tile 128x128 fp32 -> bf16 LDS (coalesced float4 + convert)
  for (int it = tid; it < 128 * CIN_ / 4; it += 256) {
    float4 v = ((const float4*)x)[(size_t)rowBase * CIN_ / 4 + it];
    int o = it * 4;
    ldsA[o + 0] = (__bf16)v.x; ldsA[o + 1] = (__bf16)v.y;
    ldsA[o + 2] = (__bf16)v.z; ldsA[o + 3] = (__bf16)v.w;
  }
  wait_async0();
  __syncthreads();

  const int lane = tid & 31, wid = tid >> 5;
  const int half = lane >> 4, l16 = lane & 15;
  const int mb = wid * 16;

  v8f acc[12];
  v8f zero = {};
#pragma unroll
  for (int t = 0; t < 12; t++) acc[t] = zero;

#pragma unroll
  for (int kk = 0; kk < 4; kk++) {
    const int kb = kk * 32;
    // A fragment (ISA 16-bit A 16x32 layout): lane = M, halves hold K groups
    const uint32_t* arow = (const uint32_t*)&ldsA[(mb + l16) * CIN_ + kb];
    v16bf afrag = make_frag(*(const uint4*)(arow + half * 4),
                            *(const uint4*)(arow + 8 + half * 4));
    // B fragment: lane = N, halves hold K groups (same as A)
#pragma unroll
    for (int t = 0; t < 12; t++) {
      const uint32_t* brow = (const uint32_t*)&ldsB[(t * 16 + l16) * CIN_ + kb];
      v16bf bfrag = make_frag(*(const uint4*)(brow + half * 4),
                              *(const uint4*)(brow + 8 + half * 4));
      acc[t] = __builtin_amdgcn_wmma_f32_16x16x32_bf16(
          false, afrag, false, bfrag, (short)0, acc[t], false, false);
    }
  }

  // epilogue: C layout lane->(m = r + 8*half, n = lane&15)
#pragma unroll
  for (int t = 0; t < 12; t++) {
#pragma unroll
    for (int r = 0; r < 8; r++) {
      int m = mb + r + 8 * half;
      int col = t * 16 + l16;
      size_t row = (size_t)(rowBase + m);
      float v = acc[t][r];
      if (col < COUT_) {
        hcat[row * CAT_ + col] = v + b_lin[col];
      } else if (col < COUT_ + M_) {
        Pin[row * M_ + (col - COUT_)] = v + b_ain[col - COUT_];
        hcat[row * CAT_ + col] = 0.f;   // agg_in accumulator init
      } else {
        Pout[row * M_ + (col - COUT_ - M_)] = v + b_aout[col - COUT_ - M_];
        hcat[row * CAT_ + col] = 0.f;   // agg_out accumulator init
      }
    }
  }
}

// ---------------- K2: COO SPMM scatter (both graphs), f32 atomics into hcat ----------------
__global__ __launch_bounds__(256) void k2_spmm(
    const int* __restrict__ in_rows, const int* __restrict__ in_cols,
    const float* __restrict__ in_vals,
    const int* __restrict__ out_rows, const int* __restrict__ out_cols,
    const float* __restrict__ out_vals,
    const float* __restrict__ Pin, const float* __restrict__ Pout,
    float* __restrict__ hcat) {
  const int tid = threadIdx.x;
  const int m = tid & 31;            // feature lane
  const int l = (tid >> 5) & 7;      // list
  const int e0 = blockIdx.x * 8;
  if (m == 0 && l == 0 && e0 + 64 < E_) {
    __builtin_prefetch(&in_rows[e0 + 64], 0, 0);
    __builtin_prefetch(&in_cols[e0 + 64], 0, 0);
    __builtin_prefetch(&in_vals[e0 + 64], 0, 0);
  }
#pragma unroll
  for (int i = 0; i < 8; i++) {
    int e = e0 + i;
    {
      int r = in_rows[e], c = in_cols[e];
      float v = in_vals[e];
      float p = Pin[((size_t)l * G_ + c) * M_ + m];
      atomicAdd(&hcat[((size_t)l * G_ + r) * CAT_ + COUT_ + m], v * p);
    }
    {
      int r = out_rows[e], c = out_cols[e];
      float v = out_vals[e];
      float p = Pout[((size_t)l * G_ + c) * M_ + m];
      atomicAdd(&hcat[((size_t)l * G_ + r) * CAT_ + COUT_ + M_ + m], v * p);
    }
  }
}

// ---------------- K4: fused silu+LN1 + MLP GEMM + bias + silu + LN2 ----------------
// block: 256 threads = 8 waves, tile 128 rows; every wave normalizes its own
// 16 rows of hcat straight into the bf16 A-tile, then wave 0 DMAs the 48KB
// weight tile into LDS via TDM, then WMMA GEMM and a
// register-only LN2 (each wave owns complete output rows).
__global__ __launch_bounds__(256, 1) void k4_fused(
    const float* __restrict__ hcat, const __bf16* __restrict__ Wmlp,
    const float* __restrict__ g1, const float* __restrict__ b1,
    const float* __restrict__ b_mlp, const float* __restrict__ g2,
    const float* __restrict__ b2, float* __restrict__ out) {
  __shared__ alignas(16) __bf16 ldsA[128 * CAT_];    // 48 KB (normalized rows)
  __shared__ alignas(16) __bf16 ldsB[CAT_ * COUT_];  // 48 KB (weights)
  const int tid = threadIdx.x;
  const int rowBase = blockIdx.x * 128;
  const int lane = tid & 31, wid = tid >> 5;
  const int half = lane >> 4, l16 = lane & 15;
  const int mb = wid * 16;

  // silu + LayerNorm(192) per row -> bf16 A-tile
  float gg[6], bbv[6];
#pragma unroll
  for (int j = 0; j < 6; j++) { gg[j] = g1[lane + 32 * j]; bbv[j] = b1[lane + 32 * j]; }
  for (int rr = 0; rr < 16; rr++) {
    const size_t row = (size_t)(rowBase + mb + rr);
    const float* hrow = hcat + row * CAT_;
    float v[6], s = 0.f, s2 = 0.f;
#pragma unroll
    for (int j = 0; j < 6; j++) {
      float xv = silu_f(hrow[lane + 32 * j]);
      v[j] = xv; s += xv; s2 += xv * xv;
    }
#pragma unroll
    for (int off = 1; off < 32; off <<= 1) {
      s += __shfl_xor(s, off, 32);
      s2 += __shfl_xor(s2, off, 32);
    }
    float mu = s * (1.f / CAT_);
    float var = s2 * (1.f / CAT_) - mu * mu;
    float inv = rsqrtf(var + EPS_);
#pragma unroll
    for (int j = 0; j < 6; j++)
      ldsA[(mb + rr) * CAT_ + lane + 32 * j] = (__bf16)((v[j] - mu) * inv * gg[j] + bbv[j]);
  }
  // weights: single TDM descriptor from wave 0 (waits TENSORcnt inside)
  if (wid == 0)
    tdm_load_to_lds(lds_off(&ldsB[0]), Wmlp, CAT_ * COUT_ * 2);
  __syncthreads();

  v8f acc[8];
  v8f zero = {};
#pragma unroll
  for (int t = 0; t < 8; t++) acc[t] = zero;

#pragma unroll
  for (int kk = 0; kk < 6; kk++) {
    const int kb = kk * 32;
    const uint32_t* arow = (const uint32_t*)&ldsA[(mb + l16) * CAT_ + kb];
    v16bf afrag = make_frag(*(const uint4*)(arow + half * 4),
                            *(const uint4*)(arow + 8 + half * 4));
#pragma unroll
    for (int t = 0; t < 8; t++) {
      const uint32_t* brow = (const uint32_t*)&ldsB[(t * 16 + l16) * CAT_ + kb];
      v16bf bfrag = make_frag(*(const uint4*)(brow + half * 4),
                              *(const uint4*)(brow + 8 + half * 4));
      acc[t] = __builtin_amdgcn_wmma_f32_16x16x32_bf16(
          false, afrag, false, bfrag, (short)0, acc[t], false, false);
    }
  }

  // bias + silu; per-row LN2 entirely in registers (wave owns full rows)
  float s1[8], s2[8];
#pragma unroll
  for (int r = 0; r < 8; r++) { s1[r] = 0.f; s2[r] = 0.f; }
#pragma unroll
  for (int t = 0; t < 8; t++) {
    float bb = b_mlp[t * 16 + l16];
#pragma unroll
    for (int r = 0; r < 8; r++) {
      float y = silu_f(acc[t][r] + bb);
      acc[t][r] = y;
      s1[r] += y; s2[r] += y * y;
    }
  }
#pragma unroll
  for (int r = 0; r < 8; r++) {
#pragma unroll
    for (int off = 1; off < 16; off <<= 1) {   // reduce within half (16 lanes = one row set)
      s1[r] += __shfl_xor(s1[r], off, 32);
      s2[r] += __shfl_xor(s2[r], off, 32);
    }
  }
#pragma unroll
  for (int r = 0; r < 8; r++) {
    float mu = s1[r] * (1.f / COUT_);
    float var = s2[r] * (1.f / COUT_) - mu * mu;
    float inv = rsqrtf(var + EPS_);
    size_t row = (size_t)(rowBase + mb + r + 8 * half);
#pragma unroll
    for (int t = 0; t < 8; t++) {
      int c = t * 16 + l16;
      out[row * COUT_ + c] = (acc[t][r] - mu) * inv * g2[c] + b2[c];
    }
  }
}

// ---------------- launcher ----------------
extern "C" void kernel_launch(void* const* d_in, const int* in_sizes, int n_in,
                              void* d_out, int out_size, void* d_ws, size_t ws_size,
                              hipStream_t stream) {
  (void)in_sizes; (void)n_in; (void)out_size; (void)ws_size;
  const float* x        = (const float*)d_in[0];
  const int*   in_rows  = (const int*)d_in[1];
  const int*   in_cols  = (const int*)d_in[2];
  const float* in_vals  = (const float*)d_in[3];
  const int*   out_rows = (const int*)d_in[4];
  const int*   out_cols = (const int*)d_in[5];
  const float* out_vals = (const float*)d_in[6];
  const float* W_lin  = (const float*)d_in[7];
  const float* b_lin  = (const float*)d_in[8];
  const float* W_ain  = (const float*)d_in[9];
  const float* b_ain  = (const float*)d_in[10];
  const float* W_aout = (const float*)d_in[11];
  const float* b_aout = (const float*)d_in[12];
  const float* ln1_g  = (const float*)d_in[13];
  const float* ln1_b  = (const float*)d_in[14];
  const float* W_mlp  = (const float*)d_in[15];
  const float* b_mlp  = (const float*)d_in[16];
  const float* ln2_g  = (const float*)d_in[17];
  const float* ln2_b  = (const float*)d_in[18];

  char* ws = (char*)d_ws;
  size_t off = 0;
  auto alloc = [&](size_t bytes) -> void* {
    void* p = ws + off;
    off = (off + bytes + 255) & ~(size_t)255;
    return p;
  };
  float*  hcat = (float*) alloc((size_t)N_ * CAT_ * 4);   // pre-activation concat
  float*  Pin  = (float*) alloc((size_t)N_ * M_ * 4);
  float*  Pout = (float*) alloc((size_t)N_ * M_ * 4);
  __bf16* Wcat = (__bf16*)alloc((size_t)CIN_ * CAT_ * 2);
  __bf16* Wm   = (__bf16*)alloc((size_t)CAT_ * COUT_ * 2);

  k0_convert<<<(CIN_ * CAT_ + 255) / 256, 256, 0, stream>>>(W_lin, W_ain, W_aout, W_mlp, Wcat, Wm);
  k1_gemm_in<<<N_ / 128, 256, 0, stream>>>(x, Wcat, b_lin, b_ain, b_aout, hcat, Pin, Pout);
  k2_spmm<<<E_ / 8, 256, 0, stream>>>(in_rows, in_cols, in_vals,
                                      out_rows, out_cols, out_vals, Pin, Pout, hcat);
  k4_fused<<<N_ / 128, 256, 0, stream>>>(hcat, Wm, ln1_g, ln1_b,
                                         b_mlp, ln2_g, ln2_b, (float*)d_out);
}
